// Decoder_Block_77738908057674
// MI455X (gfx1250) — matrix operand order repaired and checked by reading, not compiled, not run on hardware
//
#include <hip/hip_runtime.h>

// Decoder block (DrugGEN-style) for MI455X / gfx1250.
// All linears + batched edge/node aggregations run on v_wmma_f32_16x16x32_f16.

#define CC 128
#define INV_SCALE 3.3635856610148585f   // 128**0.25

typedef __attribute__((ext_vector_type(16))) _Float16 v16h;
typedef __attribute__((ext_vector_type(8)))  float    v8f;

// ---------------------------------------------------------------- fp32->fp16
__global__ __launch_bounds__(256)
void k_cvt16(const float* __restrict__ s, _Float16* __restrict__ d, long long n) {
    long long i = (long long)blockIdx.x * 256 + threadIdx.x;
    if (i < n) d[i] = (_Float16)s[i];
}

// ------------------------------------------------------- LayerNorm over 128
// one wave per row; lane handles 4 consecutive elements; optional residual.
__global__ __launch_bounds__(256)
void k_ln128(const float* __restrict__ x, const float* __restrict__ r,
             const float* __restrict__ g, const float* __restrict__ be,
             float* __restrict__ y, long long rows) {
    long long row = (long long)blockIdx.x * 8 + (threadIdx.x >> 5);
    if (row >= rows) return;
    int lane = threadIdx.x & 31;
    float4 xv = *(const float4*)(x + row * CC + lane * 4);
    if (r) {
        float4 rv = *(const float4*)(r + row * CC + lane * 4);
        xv.x += rv.x; xv.y += rv.y; xv.z += rv.z; xv.w += rv.w;
    }
    float s = xv.x + xv.y + xv.z + xv.w;
#pragma unroll
    for (int o = 16; o; o >>= 1) s += __shfl_xor(s, o, 32);
    float mean = s * (1.0f / 128.0f);
    float dx = xv.x - mean, dy = xv.y - mean, dz = xv.z - mean, dw = xv.w - mean;
    float q = dx * dx + dy * dy + dz * dz + dw * dw;
#pragma unroll
    for (int o = 16; o; o >>= 1) q += __shfl_xor(q, o, 32);
    float inv = rsqrtf(q * (1.0f / 128.0f) + 1e-5f);
    float4 gv = *(const float4*)(g + lane * 4);
    float4 bv = *(const float4*)(be + lane * 4);
    float4 yv;
    yv.x = dx * inv * gv.x + bv.x;
    yv.y = dy * inv * gv.y + bv.y;
    yv.z = dz * inv * gv.z + bv.z;
    yv.w = dw * inv * gv.w + bv.w;
    *(float4*)(y + row * CC + lane * 4) = yv;
}

// -------------------------------------------------- generic batched WMMA GEMM
// C[m,n] = epi( sum_k A[m,k]*B[k,n] + bias[n] (+resid) )
// BLAYOUT 0: B = f16 weights W[n*ldb+k] (NK), bias required, 8 waves/block span
//            128 cols, A tile (16x64) staged once in LDS for all waves.
// BLAYOUT 1: B = fp32 slabs B[k*ldb+n] (KN) with batch stride, 1 wave/block, N=16.
// K must be a multiple of 64; M,N multiples of 16 (grids divide exactly).
// EPI: 0 none, 1 sigmoid, 2 add resid[m*ldc+n].
template<int BLAYOUT, int EPI, bool SHARED_A>
__global__ __launch_bounds__(256)
void k_wmma_gemm(const float* __restrict__ A, int lda, long long bsA, int aDiv,
                 const void* __restrict__ Bsrc, int ldb, long long bsB,
                 float* __restrict__ C, int ldc, long long bsC,
                 const float* __restrict__ bias, const float* __restrict__ resid,
                 int M, int K) {
    const int wave = threadIdx.x >> 5;
    const int lane = threadIdx.x & 31;
    const int lo = lane & 15, hi = lane >> 4;
    const int nbase = blockIdx.y * (SHARED_A ? 128 : 16) + wave * 16;
    const int mbase = blockIdx.x * 16;
    const long long z = blockIdx.z;
    const float* Ab = A + (z / aDiv) * bsA + (long long)mbase * lda;
    float* Cb = C + z * bsC;
    const int col = nbase + lo;

    __shared__ _Float16 As[SHARED_A ? 16 * 64 : 8];

    v8f acc0 = {}, acc1 = {};
    for (int kb = 0; kb < K; kb += 64) {
        // ---- A fragments for two 16x32 K-blocks (ISA A layout: lanes 0-15 hold
        // row M=lane, K=hi*8..hi*8+7 in VGPR0-3 and K=16+hi*8.. in VGPR4-7)
        v16h a0, a1;
        if constexpr (SHARED_A) {
            __syncthreads();                       // WAR vs previous iteration
            const int idx = threadIdx.x * 4;       // 256 thr * 4 = 16x64 tile
            const int arow = idx >> 6, acol = idx & 63;
            float4 av = *(const float4*)(Ab + (long long)arow * lda + kb + acol);
            if (kb + 64 < K) __builtin_prefetch(Ab + (long long)arow * lda + kb + 64 + acol, 0, 1);
            _Float16* dst = &As[arow * 64 + acol];
            dst[0] = (_Float16)av.x; dst[1] = (_Float16)av.y;
            dst[2] = (_Float16)av.z; dst[3] = (_Float16)av.w;
            __syncthreads();
            const _Float16* ar = &As[lo * 64];
#pragma unroll
            for (int i = 0; i < 8; ++i) {
                a0[i]     = ar[hi * 8 + i];
                a0[8 + i] = ar[16 + hi * 8 + i];
                a1[i]     = ar[32 + hi * 8 + i];
                a1[8 + i] = ar[48 + hi * 8 + i];
            }
        } else {
            const float* ar = Ab + (long long)lo * lda + kb;
            float4 x0 = *(const float4*)(ar + hi * 8);
            float4 x1 = *(const float4*)(ar + hi * 8 + 4);
            float4 x2 = *(const float4*)(ar + 16 + hi * 8);
            float4 x3 = *(const float4*)(ar + 16 + hi * 8 + 4);
            float4 y0 = *(const float4*)(ar + 32 + hi * 8);
            float4 y1 = *(const float4*)(ar + 32 + hi * 8 + 4);
            float4 y2 = *(const float4*)(ar + 48 + hi * 8);
            float4 y3 = *(const float4*)(ar + 48 + hi * 8 + 4);
            a0[0] = (_Float16)x0.x;  a0[1] = (_Float16)x0.y;  a0[2]  = (_Float16)x0.z;  a0[3]  = (_Float16)x0.w;
            a0[4] = (_Float16)x1.x;  a0[5] = (_Float16)x1.y;  a0[6]  = (_Float16)x1.z;  a0[7]  = (_Float16)x1.w;
            a0[8] = (_Float16)x2.x;  a0[9] = (_Float16)x2.y;  a0[10] = (_Float16)x2.z;  a0[11] = (_Float16)x2.w;
            a0[12] = (_Float16)x3.x; a0[13] = (_Float16)x3.y; a0[14] = (_Float16)x3.z;  a0[15] = (_Float16)x3.w;
            a1[0] = (_Float16)y0.x;  a1[1] = (_Float16)y0.y;  a1[2]  = (_Float16)y0.z;  a1[3]  = (_Float16)y0.w;
            a1[4] = (_Float16)y1.x;  a1[5] = (_Float16)y1.y;  a1[6]  = (_Float16)y1.z;  a1[7]  = (_Float16)y1.w;
            a1[8] = (_Float16)y2.x;  a1[9] = (_Float16)y2.y;  a1[10] = (_Float16)y2.z;  a1[11] = (_Float16)y2.w;
            a1[12] = (_Float16)y3.x; a1[13] = (_Float16)y3.y; a1[14] = (_Float16)y3.z;  a1[15] = (_Float16)y3.w;
        }
        // ---- B fragments (32x16: lane holds column `col`, K = hi*8 + i and 16 + hi*8 + i)
        v16h b0, b1;
        if constexpr (BLAYOUT == 0) {
            const _Float16* wp = (const _Float16*)Bsrc + (long long)col * ldb + kb;
#pragma unroll
            for (int i = 0; i < 8; ++i) {
                b0[i]     = wp[hi * 8 + i];
                b0[8 + i] = wp[16 + hi * 8 + i];
                b1[i]     = wp[32 + hi * 8 + i];
                b1[8 + i] = wp[48 + hi * 8 + i];
            }
        } else {
            const float* bp = (const float*)Bsrc + z * bsB +
                              (long long)(kb + hi * 8) * ldb + col;
#pragma unroll
            for (int i = 0; i < 16; ++i) b0[i] = (_Float16)bp[(long long)(i + (i >> 3) * 8) * ldb];
            const float* bp1 = bp + 32LL * ldb;
#pragma unroll
            for (int i = 0; i < 16; ++i) b1[i] = (_Float16)bp1[(long long)(i + (i >> 3) * 8) * ldb];
        }
        acc0 = __builtin_amdgcn_wmma_f32_16x16x32_f16(false, a0, false, b0, (short)0, acc0, false, false);
        acc1 = __builtin_amdgcn_wmma_f32_16x16x32_f16(false, a1, false, b1, (short)0, acc1, false, false);
    }
    const float bvv = (BLAYOUT == 0) ? bias[col] : 0.0f;
#pragma unroll
    for (int rr = 0; rr < 8; ++rr) {
        const int m = mbase + rr + 8 * hi;
        float v = acc0[rr] + acc1[rr] + bvv;
        if constexpr (EPI == 1) v = 1.0f / (1.0f + __expf(-v));
        if constexpr (EPI == 2) v += resid[(long long)m * ldc + col];
        Cb[(long long)m * ldc + col] = v;
    }
}

// ------------------------------------------------ prot scores s[bh,m,n] (d=16)
__global__ __launch_bounds__(256)
void k_score_prot(const float* __restrict__ q, const float* __restrict__ kk, float* __restrict__ s) {
    const int blk = blockIdx.x;          // bh*256 + m   (16384 blocks)
    const int n = threadIdx.x;
    const float* qv = q + (long long)blk * 16;
    const float* kv = kk + (long long)blk * 4096 + n * 16;
    float acc = 0.f;
#pragma unroll
    for (int d = 0; d < 16; ++d) acc += qv[d] * kv[d];
    s[(long long)blk * 256 + n] = acc * INV_SCALE;
}

// ---------------------------------------- softmax stats (max + sumexp) per row
// row base = (r/inner)*ostride + (r%inner)*istride ; elems at base + i*stride.
__global__ __launch_bounds__(256)
void k_stats(const float* __restrict__ x, long long inner, long long ostride,
             long long istride, long long stride, int count,
             float* __restrict__ omax, float* __restrict__ osum) {
    const long long r = blockIdx.x;
    const float* p = x + (r / inner) * ostride + (r % inner) * istride;
    float mx = -3.4e38f;
    for (int i = threadIdx.x; i < count; i += 256) mx = fmaxf(mx, p[(long long)i * stride]);
    __shared__ float sm[8];
#pragma unroll
    for (int o = 16; o; o >>= 1) mx = fmaxf(mx, __shfl_xor(mx, o, 32));
    if ((threadIdx.x & 31) == 0) sm[threadIdx.x >> 5] = mx;
    __syncthreads();
    float bmax = sm[0];
#pragma unroll
    for (int i = 1; i < 8; ++i) bmax = fmaxf(bmax, sm[i]);
    __syncthreads();
    float ss = 0.f;
    for (int i = threadIdx.x; i < count; i += 256) ss += __expf(p[(long long)i * stride] - bmax);
#pragma unroll
    for (int o = 16; o; o >>= 1) ss += __shfl_xor(ss, o, 32);
    if ((threadIdx.x & 31) == 0) sm[threadIdx.x >> 5] = ss;
    __syncthreads();
    if (threadIdx.x == 0) {
        float t = 0.f;
#pragma unroll
        for (int i = 0; i < 8; ++i) t += sm[i];
        omax[r] = bmax; osum[r] = t;
    }
}

// message = out_attn + in_attn(transposed read) - eye*out_attn   (prot stage)
__global__ __launch_bounds__(256)
void k_msg_prot(const float* __restrict__ s, const float* __restrict__ rmax,
                const float* __restrict__ rsum, const float* __restrict__ cmax,
                const float* __restrict__ csum, float* __restrict__ msg) {
    long long idx = (long long)blockIdx.x * 256 + threadIdx.x;   // 16777216
    int n = (int)(idx & 255);
    long long r = idx >> 8;              // bh*256 + m
    int m = (int)(r & 255);
    long long bh = r >> 8;
    float outv = __expf(s[idx] - rmax[r]) / rsum[r];
    long long ci = (bh << 8) + m;
    float inv = __expf(s[((bh << 8) + n) * 256 + m] - cmax[ci]) / csum[ci];
    float v = outv + inv;
    if (m == n) v -= outv;
    msg[idx] = v;
}

// out[m,n] = rowsoftmax(s)[m,n] + colsoftmax(s)[m,n]   (enc-dec messages)
__global__ __launch_bounds__(256)
void k_msg_dual(const float* __restrict__ s, const float* __restrict__ rmax,
                const float* __restrict__ rsum, const float* __restrict__ cmax,
                const float* __restrict__ csum, float* __restrict__ out,
                int Mrows, int Ncols, long long total) {
    long long idx = (long long)blockIdx.x * 256 + threadIdx.x;
    if (idx >= total) return;
    int n = (int)(idx % Ncols);
    long long r = idx / Ncols;           // bh*Mrows + m
    long long bh = r / Mrows;
    long long ci = bh * Ncols + n;
    float sv = s[idx];
    out[idx] = __expf(sv - rmax[r]) / rsum[r] + __expf(sv - cmax[ci]) / csum[ci];
}

// enc-dec scores s2[bh,m(64),n(256)]
__global__ __launch_bounds__(256)
void k_score_x(const float* __restrict__ qx, const float* __restrict__ kx, float* __restrict__ s2) {
    int idx = blockIdx.x * 256 + threadIdx.x;        // 1048576
    int n = idx & 255;
    int bhm = idx >> 8;                              // bh*64 + m
    int bh = bhm >> 6;
    const float* qv = qx + (long long)bhm * 16;
    const float* kv = kx + ((long long)bh * 256 + n) * 16;
    float acc = 0.f;
#pragma unroll
    for (int d = 0; d < 16; ++d) acc += qv[d] * kv[d];
    s2[idx] = acc * INV_SCALE;
}

__global__ __launch_bounds__(256)
void k_rowsum(const float* __restrict__ x, float* __restrict__ out, int rows, int cols) {
    int r = blockIdx.x * 256 + threadIdx.x;
    if (r >= rows) return;
    const float* p = x + (long long)r * cols;
    float s = 0.f;
    for (int i = 0; i < cols; ++i) s += p[i];
    out[r] = s;
}

// node_h = rowsum(message) * vx  (flat layouts coincide)
__global__ __launch_bounds__(256)
void k_node_x(const float* __restrict__ rs, const float* __restrict__ vx, float* __restrict__ nh) {
    int idx = blockIdx.x * 256 + threadIdx.x;        // 65536
    nh[idx] = rs[idx >> 4] * vx[idx];
}

// ksum[bh,d] = sum_{m,n} ka[bh,m,n,d]
__global__ __launch_bounds__(256)
void k_ksum(const float* __restrict__ ka, float* __restrict__ ks) {
    const int bh = blockIdx.x;                       // 64
    float acc[16];
#pragma unroll
    for (int d = 0; d < 16; ++d) acc[d] = 0.f;
    const float* base = ka + (long long)bh * (256LL * 256 * 16) + (long long)threadIdx.x * 16;
    for (int m = 0; m < 256; ++m) {
        const float* p = base + (long long)m * 4096;
#pragma unroll
        for (int d = 0; d < 16; ++d) acc[d] += p[d];
    }
    __shared__ float sm[256 * 16];
#pragma unroll
    for (int d = 0; d < 16; ++d) sm[threadIdx.x * 16 + d] = acc[d];
    __syncthreads();
    if (threadIdx.x < 16) {
        float t = 0.f;
        for (int i = 0; i < 256; ++i) t += sm[i * 16 + threadIdx.x];
        ks[bh * 16 + threadIdx.x] = t;
    }
}

// se[bh,m,n] = INV * dot(qa[bh,m,n,:], ksum[bh,:])
__global__ __launch_bounds__(256)
void k_score_se(const float* __restrict__ qa, const float* __restrict__ ks, float* __restrict__ se) {
    int idx = blockIdx.x * 256 + threadIdx.x;        // 262144
    int bh = idx >> 12;
    const float* qv = qa + (long long)idx * 16;
    const float* kv = ks + bh * 16;
    float acc = 0.f;
#pragma unroll
    for (int d = 0; d < 16; ++d) acc += qv[d] * kv[d];
    se[idx] = acc * INV_SCALE;
}

// vasum[bh,m,d] = sum_n va[bh,m,n,d]
__global__ __launch_bounds__(256)
void k_vasum(const float* __restrict__ va, float* __restrict__ vs) {
    int idx = blockIdx.x * 256 + threadIdx.x;        // 65536
    int d = idx & 15;
    int bhm = idx >> 4;
    const float* base = va + (long long)bhm * 1024 + d;
    float s = 0.f;
    for (int n = 0; n < 64; ++n) s += base[n * 16];
    vs[idx] = s;
}

// edge_h[bh,m,n,d] = me2[bh,m,n] * vasum[bh,m,d]
__global__ __launch_bounds__(256)
void k_edge2(const float* __restrict__ me2, const float* __restrict__ vs, float* __restrict__ out) {
    long long idx = (long long)blockIdx.x * 256 + threadIdx.x;   // 8388608
    int d = (int)(idx & 15);
    long long rest = idx >> 4;            // bhm*64 + n
    long long bhm = rest >> 6;
    out[idx] = me2[rest] * vs[bhm * 16 + d];
}

// ---------------------------------------------------------------- host helpers
static inline void gemm_lin(hipStream_t st, const float* A, int lda, const _Float16* W,
                            float* C, int ldc, const float* bias, const float* resid,
                            int M, int N, int K, int epi) {
    dim3 grid(M / 16, N / 128, 1);
    if (epi == 0)
        k_wmma_gemm<0, 0, true><<<grid, 256, 0, st>>>(A, lda, 0, 1, W, K, 0, C, ldc, 0, bias, resid, M, K);
    else if (epi == 1)
        k_wmma_gemm<0, 1, true><<<grid, 256, 0, st>>>(A, lda, 0, 1, W, K, 0, C, ldc, 0, bias, resid, M, K);
    else
        k_wmma_gemm<0, 2, true><<<grid, 256, 0, st>>>(A, lda, 0, 1, W, K, 0, C, ldc, 0, bias, resid, M, K);
}
static inline void gemm_bat(hipStream_t st, const float* A, int lda, long long bsA, int aDiv,
                            const float* B, int ldb, long long bsB,
                            float* C, int ldc, long long bsC, int M, int K, int batches) {
    dim3 grid(M / 16, 1, batches);
    k_wmma_gemm<1, 0, false><<<grid, 32, 0, st>>>(A, lda, bsA, aDiv, B, ldb, bsB,
                                                  C, ldc, bsC, nullptr, nullptr, M, K);
}

extern "C" void kernel_launch(void* const* d_in, const int* in_sizes, int n_in,
                              void* d_out, int out_size, void* d_ws, size_t ws_size,
                              hipStream_t stream) {
    (void)in_sizes; (void)n_in; (void)out_size; (void)ws_size;
    const float* mol_annot  = (const float*)d_in[0];    // [8,64,128]
    const float* prot_annot = (const float*)d_in[1];    // [8,256,128]
    const float* mol_adj    = (const float*)d_in[2];    // [8,64,64,128]
    const float* prot_adj   = (const float*)d_in[3];    // [8,256,256,128]
    const float* att_w  = (const float*)d_in[4];        // [5,128,128]
    const float* att_b  = (const float*)d_in[5];
    const float* dec_w  = (const float*)d_in[6];        // [8,128,128]
    const float* dec_b  = (const float*)d_in[7];
    const float* mlp_w1 = (const float*)d_in[8];        // [2,512,128]
    const float* mlp_b1 = (const float*)d_in[9];
    const float* mlp_w2 = (const float*)d_in[10];       // [2,128,512]
    const float* mlp_b2 = (const float*)d_in[11];
    const float* ln_g   = (const float*)d_in[12];       // [10,128]
    const float* ln_b   = (const float*)d_in[13];

    float* out_edge = (float*)d_out;                    // [8,64,64,128]
    float* out_node = out_edge + 4194304;               // [8,64,128]
    float* out_attn = out_node + 65536;                 // [8,8,64,256]

    // -------- workspace layout
    _Float16* w16_att = (_Float16*)d_ws;                // 81920
    _Float16* w16_dec = w16_att + 81920;                // 131072
    _Float16* w16_m1  = w16_dec + 131072;               // 131072
    _Float16* w16_m2  = w16_m1  + 131072;               // 131072
    float* f = (float*)(w16_m2 + 131072);
    float* bigA  = f; f += 67108864;    // pa_ln
    float* bigB  = f; f += 67108864;    // kk, later pa1
    float* bigC  = f; f += 67108864;    // edge1 rows, later ka
    float* bigD  = f; f += 67108864;    // pa1_pre
    float* ma_ln = f; f += 4194304;
    float* mx_ln = f; f += 65536;
    float* px_ln = f; f += 262144;
    float* q_    = f; f += 262144;
    float* v_    = f; f += 262144;
    float* s1    = f; f += 4194304;
    float* msg1  = f; f += 4194304;
    float* rmax  = f; f += 16384;
    float* rsum  = f; f += 16384;
    float* cmax  = f; f += 16384;
    float* csum  = f; f += 16384;
    float* nh1   = f; f += 262144;
    float* px1p  = f; f += 262144;
    float* px1   = f; f += 262144;
    float* qx_   = f; f += 65536;
    float* kx_   = f; f += 262144;
    float* vx_   = f; f += 65536;
    float* qa_   = f; f += 4194304;
    float* va_   = f; f += 4194304;
    float* s2    = f; f += 1048576;
    float* ksb   = f; f += 1024;
    float* seb   = f; f += 262144;
    float* me2   = f; f += 262144;
    float* vs    = f; f += 65536;
    float* rs    = f; f += 4096;
    float* nh2   = f; f += 65536;
    float* mx1p  = f; f += 65536;
    float* mx1   = f; f += 65536;
    float* edge2 = f; f += 4194304;
    float* ma1p  = f; f += 4194304;
    float* ma1   = f; f += 4194304;
    float* h1_   = f; f += 16777216;
    float* ma2_  = f; f += 4194304;
    float* hx1_  = f; f += 262144;
    float* mx2_  = f; f += 65536;

    const long long BIG = 1LL << 40;

    // -------- 0. weights -> f16
    k_cvt16<<<(81920 + 255) / 256, 256, 0, stream>>>(att_w, w16_att, 81920);
    k_cvt16<<<(131072 + 255) / 256, 256, 0, stream>>>(dec_w, w16_dec, 131072);
    k_cvt16<<<(131072 + 255) / 256, 256, 0, stream>>>(mlp_w1, w16_m1, 131072);
    k_cvt16<<<(131072 + 255) / 256, 256, 0, stream>>>(mlp_w2, w16_m2, 131072);

    // -------- 1. input LayerNorms
    k_ln128<<<32768 / 8, 256, 0, stream>>>(mol_adj,   nullptr, ln_g + 0 * 128, ln_b + 0 * 128, ma_ln, 32768);
    k_ln128<<<524288 / 8, 256, 0, stream>>>(prot_adj, nullptr, ln_g + 1 * 128, ln_b + 1 * 128, bigA, 524288);
    k_ln128<<<512 / 8, 256, 0, stream>>>(mol_annot,   nullptr, ln_g + 2 * 128, ln_b + 2 * 128, mx_ln, 512);
    k_ln128<<<2048 / 8, 256, 0, stream>>>(prot_annot, nullptr, ln_g + 3 * 128, ln_b + 3 * 128, px_ln, 2048);

    // -------- 2. _attention_new(px, pa)
    gemm_lin(stream, px_ln, 128, w16_att,             q_,   128, att_b,           nullptr, 2048,   128, 128, 0);
    gemm_lin(stream, bigA,  128, w16_att + 16384,     bigB, 128, att_b + 128,     nullptr, 524288, 128, 128, 0);
    gemm_lin(stream, px_ln, 128, w16_att + 2 * 16384, v_,   128, att_b + 2 * 128, nullptr, 2048,   128, 128, 0);
    k_score_prot<<<16384, 256, 0, stream>>>(q_, bigB, s1);
    k_stats<<<16384, 256, 0, stream>>>(s1, BIG, 0, 256, 1, 256, rmax, rsum);      // row softmax stats
    k_stats<<<16384, 256, 0, stream>>>(s1, 256, 65536, 1, 256, 256, cmax, csum);  // col softmax stats
    k_msg_prot<<<65536, 256, 0, stream>>>(s1, rmax, rsum, cmax, csum, msg1);
    // edge_h: 16384 batched [256,256]@[256,16] (WMMA); k-slabs are contiguous in flat view
    gemm_bat(stream, msg1, 256, 65536, 256, bigB, 16, 4096, bigC, 16, 4096, 256, 256, 16384);
    // node_h: 64 batched [256,256]@[256,16]
    gemm_bat(stream, msg1, 256, 65536, 1,   v_,   16, 4096, nh1,  16, 4096, 256, 256, 64);
    gemm_lin(stream, nh1, 128, w16_att + 4 * 16384, px1p, 128, att_b + 4 * 128, nullptr, 2048, 128, 128, 0);
    k_ln128<<<2048 / 8, 256, 0, stream>>>(px1p, px_ln, ln_g + 5 * 128, ln_b + 5 * 128, px1, 2048);
    gemm_lin(stream, bigC, 128, w16_att + 3 * 16384, bigD, 128, att_b + 3 * 128, nullptr, 524288, 128, 128, 0);
    k_ln128<<<524288 / 8, 256, 0, stream>>>(bigD, bigA, ln_g + 4 * 128, ln_b + 4 * 128, bigB, 524288);   // pa1 -> bigB

    // -------- 3. _enc_dec_attention(mx, px1, ma, pa1)
    gemm_lin(stream, mx_ln, 128, w16_dec,             qx_,  128, dec_b,           nullptr, 512,    128, 128, 0);
    gemm_lin(stream, px1,   128, w16_dec + 16384,     kx_,  128, dec_b + 128,     nullptr, 2048,   128, 128, 0);
    gemm_lin(stream, mx_ln, 128, w16_dec + 2 * 16384, vx_,  128, dec_b + 2 * 128, nullptr, 512,    128, 128, 0);
    gemm_lin(stream, ma_ln, 128, w16_dec + 3 * 16384, qa_,  128, dec_b + 3 * 128, nullptr, 32768,  128, 128, 0);
    gemm_lin(stream, bigB,  128, w16_dec + 4 * 16384, bigC, 128, dec_b + 4 * 128, nullptr, 524288, 128, 128, 0); // ka
    gemm_lin(stream, ma_ln, 128, w16_dec + 5 * 16384, va_,  128, dec_b + 5 * 128, nullptr, 32768,  128, 128, 0);
    k_score_x<<<4096, 256, 0, stream>>>(qx_, kx_, s2);
    k_stats<<<4096, 256, 0, stream>>>(s2, BIG, 0, 256, 1, 256, rmax, rsum);
    k_stats<<<16384, 256, 0, stream>>>(s2, 256, 16384, 1, 256, 64, cmax, csum);
    k_msg_dual<<<4096, 256, 0, stream>>>(s2, rmax, rsum, cmax, csum, out_attn, 64, 256, 1048576);  // attn_dec
    k_rowsum<<<16, 256, 0, stream>>>(out_attn, rs, 4096, 256);
    k_node_x<<<256, 256, 0, stream>>>(rs, vx_, nh2);
    k_ksum<<<64, 256, 0, stream>>>(bigC, ksb);
    k_score_se<<<1024, 256, 0, stream>>>(qa_, ksb, seb);
    k_stats<<<4096, 256, 0, stream>>>(seb, BIG, 0, 64, 1, 64, rmax, rsum);
    k_stats<<<4096, 256, 0, stream>>>(seb, 64, 4096, 1, 64, 64, cmax, csum);
    k_msg_dual<<<1024, 256, 0, stream>>>(seb, rmax, rsum, cmax, csum, me2, 64, 64, 262144);
    k_vasum<<<256, 256, 0, stream>>>(va_, vs);
    k_edge2<<<32768, 256, 0, stream>>>(me2, vs, edge2);
    gemm_lin(stream, nh2,   128, w16_dec + 6 * 16384, mx1p, 128, dec_b + 6 * 128, nullptr, 512,   128, 128, 0);
    k_ln128<<<512 / 8, 256, 0, stream>>>(mx1p, mx_ln, ln_g + 7 * 128, ln_b + 7 * 128, mx1, 512);
    gemm_lin(stream, edge2, 128, w16_dec + 7 * 16384, ma1p, 128, dec_b + 7 * 128, nullptr, 32768, 128, 128, 0);
    k_ln128<<<32768 / 8, 256, 0, stream>>>(ma1p, ma_ln, ln_g + 6 * 128, ln_b + 6 * 128, ma1, 32768);

    // -------- 4. MLPs (sigmoid + residual fused into GEMM epilogues)
    gemm_lin(stream, ma1,  128, w16_m1,         h1_,  512, mlp_b1,       nullptr, 32768, 512, 128, 1);
    gemm_lin(stream, h1_,  512, w16_m2,         ma2_, 128, mlp_b2,       ma1,     32768, 128, 512, 2);
    gemm_lin(stream, mx1,  128, w16_m1 + 65536, hx1_, 512, mlp_b1 + 512, nullptr, 512,   512, 128, 1);
    gemm_lin(stream, hx1_, 512, w16_m2 + 65536, mx2_, 128, mlp_b2 + 128, mx1,     512,   128, 512, 2);

    // -------- 5. final LayerNorms straight into d_out
    k_ln128<<<32768 / 8, 256, 0, stream>>>(ma2_, nullptr, ln_g + 8 * 128, ln_b + 8 * 128, out_edge, 32768);
    k_ln128<<<512 / 8, 256, 0, stream>>>(mx2_,  nullptr, ln_g + 9 * 128, ln_b + 9 * 128, out_node, 512);
}
